// MultiHeadAttention_66760971649428
// MI455X (gfx1250) — matrix operand order repaired and checked by reading, not compiled, not run on hardware
//
#include <hip/hip_runtime.h>

typedef __attribute__((ext_vector_type(16))) __bf16 v16bf;
typedef __attribute__((ext_vector_type(8)))  __bf16 v8bf;
typedef __attribute__((ext_vector_type(8)))  float  v8f;

// Problem dims (fixed by the reference)
constexpr int Bq  = 4;
constexpr int Lq  = 2048;
constexpr int Dq  = 1024;
constexpr int Hq  = 16;
constexpr int DVq = 64;
constexpr int Mrows = Bq * Lq;   // 8192

// ---------- helpers ----------
__device__ __forceinline__ unsigned short f2bf(float f) {
  unsigned int u = __float_as_uint(f);
  u += 0x7FFFu + ((u >> 16) & 1u);      // round-to-nearest-even
  return (unsigned short)(u >> 16);
}

__device__ __forceinline__ v8bf ld8(const unsigned short* p) {
  return *(const v8bf*)p;               // 16B global_load_b128
}

__device__ __forceinline__ v16bf cat8(v8bf lo, v8bf hi) {
  return __builtin_shufflevector(lo, hi, 0,1,2,3,4,5,6,7,8,9,10,11,12,13,14,15);
}

// ---------- fp32 -> bf16 conversion (vectorized) ----------
__global__ void cvt_f32_bf16(const float* __restrict__ in,
                             unsigned short* __restrict__ out, int n4) {
  int i = blockIdx.x * blockDim.x + threadIdx.x;
  if (i < n4) {
    float4 v = ((const float4*)in)[i];
    uint2 o;
    o.x = (unsigned)f2bf(v.x) | ((unsigned)f2bf(v.y) << 16);
    o.y = (unsigned)f2bf(v.z) | ((unsigned)f2bf(v.w) << 16);
    ((uint2*)out)[i] = o;
  }
}

// ---------- GEMM: out = A(bf16)[M,K] @ W(bf16)[N,K]^T + bias ----------
// Each wave: 32x64 C tile (two 16-row A chunks share every B tile).
// MODE 0: store bf16 row-major [M,N]
// MODE 1: store bf16 transposed V layout Vt[B,H,DV,L]
// MODE 2: store f32 row-major [M,N]
template <int MODE>
__global__ __launch_bounds__(128) void gemm_xwT(
    const unsigned short* __restrict__ A,
    const unsigned short* __restrict__ W,
    const float* __restrict__ bias,
    unsigned short* __restrict__ outB,
    float* __restrict__ outF) {
  constexpr int K = Dq, N = Dq;
  const int lane = threadIdx.x & 31;
  const int wave = threadIdx.x >> 5;
  const int mBase = blockIdx.x * 128 + wave * 32;
  const int nBase = blockIdx.y * 64;
  const int mL   = lane & 15;     // A row / B column / C column
  const int half = lane >> 4;
  const int da = 8 * half;        // A-matrix k offsets: da, da+16
  const int kb = 8 * half;        // B-matrix k offsets: kb, kb+16

  v8f acc[2][4] = {};
  const unsigned short* aRow0 = A + (size_t)(mBase + mL) * K;
  const unsigned short* aRow1 = A + (size_t)(mBase + 16 + mL) * K;

  for (int kc = 0; kc < K; kc += 32) {
    v16bf a0 = cat8(ld8(aRow0 + kc + da), ld8(aRow0 + kc + da + 16));
    v16bf a1 = cat8(ld8(aRow1 + kc + da), ld8(aRow1 + kc + da + 16));
    #pragma unroll
    for (int t = 0; t < 4; ++t) {
      const unsigned short* wRow =
          W + (size_t)(nBase + t * 16 + mL) * K + kc + kb;
      v16bf b = cat8(ld8(wRow), ld8(wRow + 16));
      acc[0][t] = __builtin_amdgcn_wmma_f32_16x16x32_bf16(
          false, a0, false, b, (short)0, acc[0][t], false, false);
      acc[1][t] = __builtin_amdgcn_wmma_f32_16x16x32_bf16(
          false, a1, false, b, (short)0, acc[1][t], false, false);
    }
  }

  #pragma unroll
  for (int mt = 0; mt < 2; ++mt) {
    #pragma unroll
    for (int t = 0; t < 4; ++t) {
      int col = nBase + t * 16 + mL;
      float bv = bias[col];
      #pragma unroll
      for (int r = 0; r < 8; ++r) {
        int mRow = mBase + mt * 16 + r + 8 * half;   // C: M = r + 8*half
        float val = acc[mt][t][r] + bv;
        if (MODE == 0) {
          outB[(size_t)mRow * N + col] = f2bf(val);
        } else if (MODE == 1) {
          int bI = mRow >> 11;                 // / L
          int key = mRow & (Lq - 1);
          int h = col >> 6, dv = col & 63;
          outB[(((size_t)bI * Hq + h) * DVq + dv) * Lq + key] = f2bf(val);
        } else {
          outF[(size_t)mRow * N + col] = val;
        }
      }
    }
  }
}

// ---------- Flash attention: per wave a 16-row Q tile, 64 keys/step ----------
__global__ __launch_bounds__(128) void flash_attn(
    const unsigned short* __restrict__ Qp,   // [B,L,D] bf16
    const unsigned short* __restrict__ Kp,   // [B,L,D] bf16
    const unsigned short* __restrict__ Vt,   // [B,H,DV,L] bf16
    unsigned short* __restrict__ O) {        // [B,L,D] bf16
  // per-wave P slab (16x64 bf16); row stride 72 halfs (144B): 16B-aligned,
  // bank pattern 36*m mod 64 distinct for all 16 rows -> conflict-free b128.
  __shared__ __align__(16) unsigned short ldsP[4][16][72];

  const int lane = threadIdx.x & 31;
  const int wave = threadIdx.x >> 5;
  constexpr int nQB = Lq / 64;            // 32 q-blocks per (b,h)
  const int bh   = blockIdx.x / nQB;
  const int qblk = blockIdx.x % nQB;
  const int b = bh / Hq, h = bh % Hq;
  const int qBase = qblk * 64 + wave * 16;
  const int mL   = lane & 15;
  const int half = lane >> 4;
  const int da = 8 * half;                // A-matrix k offsets
  const int kb = 8 * half;                // B-matrix k offsets
  // log2-domain softmax: fold 1/sqrt(DV) * log2(e) into one constant
  const float sc = 0.125f * 1.44269504f;

  // Q tile resident in registers (A-layout, two k-chunks of 32)
  const unsigned short* qRow =
      Qp + (size_t)(b * Lq + qBase + mL) * Dq + h * DVq;
  v16bf qA[2];
  #pragma unroll
  for (int c = 0; c < 2; ++c)
    qA[c] = cat8(ld8(qRow + c * 32 + da), ld8(qRow + c * 32 + da + 16));

  // all-ones B matrix for the l = P*1 accumulator
  unsigned short oneBits = 0x3F80;        // bf16 1.0
  __bf16 oneBf = *(__bf16*)&oneBits;
  v16bf onesB;
  #pragma unroll
  for (int i = 0; i < 16; ++i) onesB[i] = oneBf;

  v8f oAcc[4] = {};
  v8f lAcc = {};
  float mRow[8];
  #pragma unroll
  for (int r = 0; r < 8; ++r) mRow[r] = -1e30f;

  const unsigned short* kBase = Kp + (size_t)(b * Lq) * Dq + h * DVq;
  const unsigned short* vBase = Vt + ((size_t)(b * Hq + h)) * DVq * Lq;

  for (int jb = 0; jb < Lq; jb += 64) {
    // prefetch next tiles (speculative; safe)
    __builtin_prefetch(kBase + (size_t)(jb + 64 + mL) * Dq, 0, 1);
    __builtin_prefetch(vBase + (size_t)mL * Lq + jb + 64, 0, 1);

    // --- scores S = Q @ K^T for 64 keys (four 16x16 tiles) ---
    v8f s[4];
    #pragma unroll
    for (int t = 0; t < 4; ++t) {
      v8f st = {};
      const unsigned short* kRow = kBase + (size_t)(jb + t * 16 + mL) * Dq;
      #pragma unroll
      for (int c = 0; c < 2; ++c) {
        v16bf kB = cat8(ld8(kRow + c * 32 + kb), ld8(kRow + c * 32 + kb + 16));
        st = __builtin_amdgcn_wmma_f32_16x16x32_bf16(
            false, qA[c], false, kB, (short)0, st, false, false);
      }
      s[t] = st;
    }

    // --- online softmax, log2 domain; rows span a 16-lane half ---
    float p[4][8], alpha[8];
    #pragma unroll
    for (int r = 0; r < 8; ++r) {
      float s0 = s[0][r] * sc, s1 = s[1][r] * sc;
      float s2 = s[2][r] * sc, s3 = s[3][r] * sc;
      float tm = fmaxf(fmaxf(s0, s1), fmaxf(s2, s3));
      tm = fmaxf(tm, __shfl_xor(tm, 1, 32));
      tm = fmaxf(tm, __shfl_xor(tm, 2, 32));
      tm = fmaxf(tm, __shfl_xor(tm, 4, 32));
      tm = fmaxf(tm, __shfl_xor(tm, 8, 32));
      float mNew = fmaxf(mRow[r], tm);
      alpha[r] = exp2f(mRow[r] - mNew);
      p[0][r] = exp2f(s0 - mNew);
      p[1][r] = exp2f(s1 - mNew);
      p[2][r] = exp2f(s2 - mNew);
      p[3][r] = exp2f(s3 - mNew);
      mRow[r] = mNew;
    }
    #pragma unroll
    for (int t = 0; t < 4; ++t)
      #pragma unroll
      for (int r = 0; r < 8; ++r) oAcc[t][r] *= alpha[r];
    #pragma unroll
    for (int r = 0; r < 8; ++r) lAcc[r] *= alpha[r];

    // --- re-layout P: C-format -> A-format via per-wave LDS slab ---
    #pragma unroll
    for (int r = 0; r < 8; ++r) {
      int row = r + 8 * half;
      #pragma unroll
      for (int t = 0; t < 4; ++t)
        ldsP[wave][row][mL + 16 * t] = f2bf(p[t][r]);
    }
    v16bf pA[2];
    pA[0] = cat8(*(const v8bf*)&ldsP[wave][mL][da],
                 *(const v8bf*)&ldsP[wave][mL][da + 16]);
    pA[1] = cat8(*(const v8bf*)&ldsP[wave][mL][32 + da],
                 *(const v8bf*)&ldsP[wave][mL][32 + da + 16]);

    // --- O += P @ V   (Vt rows are contiguous B-columns) ---
    #pragma unroll
    for (int t = 0; t < 4; ++t) {
      const unsigned short* vRow = vBase + (size_t)(t * 16 + mL) * Lq + jb + kb;
      #pragma unroll
      for (int c = 0; c < 2; ++c) {
        v16bf vB = cat8(ld8(vRow + c * 32), ld8(vRow + c * 32 + 16));
        oAcc[t] = __builtin_amdgcn_wmma_f32_16x16x32_bf16(
            false, pA[c], false, vB, (short)0, oAcc[t], false, false);
      }
    }
    // --- l += P @ ones  (kills the row-sum shuffle reduction) ---
    #pragma unroll
    for (int c = 0; c < 2; ++c)
      lAcc = __builtin_amdgcn_wmma_f32_16x16x32_bf16(
          false, pA[c], false, onesB, (short)0, lAcc, false, false);
  }

  // --- normalize and store merged [B,L,D] bf16 ---
  float rinv[8];
  #pragma unroll
  for (int r = 0; r < 8; ++r) rinv[r] = 1.0f / lAcc[r];
  #pragma unroll
  for (int t = 0; t < 4; ++t) {
    int col = h * DVq + t * 16 + mL;
    #pragma unroll
    for (int r = 0; r < 8; ++r) {
      int mIdx = qBase + r + 8 * half;
      O[(size_t)(b * Lq + mIdx) * Dq + col] = f2bf(oAcc[t][r] * rinv[r]);
    }
  }
}

// ---------- host launcher ----------
extern "C" void kernel_launch(void* const* d_in, const int* in_sizes, int n_in,
                              void* d_out, int out_size, void* d_ws, size_t ws_size,
                              hipStream_t stream) {
  (void)in_sizes; (void)n_in; (void)out_size; (void)ws_size;
  const float* q  = (const float*)d_in[0];
  const float* k  = (const float*)d_in[1];
  const float* v  = (const float*)d_in[2];
  const float* Wq = (const float*)d_in[3];
  const float* bq = (const float*)d_in[4];
  const float* Wk = (const float*)d_in[5];
  const float* bk = (const float*)d_in[6];
  const float* Wv = (const float*)d_in[7];
  const float* bv = (const float*)d_in[8];
  const float* Wc = (const float*)d_in[9];
  const float* bc = (const float*)d_in[10];
  float* out = (float*)d_out;

  const size_t nX = (size_t)Mrows * Dq;   // 8M activation elems
  const size_t nW = (size_t)Dq * Dq;      // 1M weight elems

  char* ws = (char*)d_ws;
  size_t off = 0;
  auto alloc = [&](size_t bytes) {
    char* p = ws + off;
    off += (bytes + 255) & ~(size_t)255;
    return (unsigned short*)p;
  };
  unsigned short* qbf  = alloc(nX * 2);
  unsigned short* kbf  = alloc(nX * 2);
  unsigned short* vbf  = alloc(nX * 2);
  unsigned short* Wqb  = alloc(nW * 2);
  unsigned short* Wkb  = alloc(nW * 2);
  unsigned short* Wvb  = alloc(nW * 2);
  unsigned short* Wcb  = alloc(nW * 2);
  unsigned short* Qp   = alloc(nX * 2);
  unsigned short* Kp   = alloc(nX * 2);
  unsigned short* VtT  = alloc(nX * 2);
  unsigned short* Obf  = qbf;   // reuse: qbf is dead after the Q projection

  auto cvt = [&](const float* src, unsigned short* dst, size_t n) {
    int n4 = (int)(n / 4);
    cvt_f32_bf16<<<dim3((n4 + 255) / 256), dim3(256), 0, stream>>>(src, dst, n4);
  };
  cvt(q, qbf, nX);
  cvt(k, kbf, nX);
  cvt(v, vbf, nX);
  cvt(Wq, Wqb, nW);
  cvt(Wk, Wkb, nW);
  cvt(Wv, Wvb, nW);
  cvt(Wc, Wcb, nW);

  dim3 gg(Mrows / 128, Dq / 64);   // (64, 16)
  gemm_xwT<0><<<gg, 128, 0, stream>>>(qbf, Wqb, bq, Qp, nullptr);
  gemm_xwT<0><<<gg, 128, 0, stream>>>(kbf, Wkb, bk, Kp, nullptr);
  gemm_xwT<1><<<gg, 128, 0, stream>>>(vbf, Wvb, bv, VtT, nullptr);

  flash_attn<<<dim3(Bq * Hq * (Lq / 64)), 128, 0, stream>>>(Qp, Kp, VtT, Obf);

  gemm_xwT<2><<<gg, 128, 0, stream>>>(Obf, Wcb, bc, nullptr, out);
}
